// QLoRALinear_67765993996313
// MI455X (gfx1250) — matrix operand order repaired and checked by reading, not compiled, not run on hardware
//
#include <hip/hip_runtime.h>
#include <stdint.h>

typedef __attribute__((ext_vector_type(16))) __bf16 v16bf;
typedef __attribute__((ext_vector_type(8)))  float  v8f;

#define IN_F   8192
#define OUT_F  8192
#define BATCH  64
#define RRANK  16
#define GROUP  16
#define KC     256                 // K elements dequantized per chunk
#define LDSW   (KC + 8)            // padded LDS row stride (bf16 elements)

__device__ __forceinline__ uint32_t f2bf_rne(float x) {
    uint32_t u = __float_as_uint(x);
    u += 0x7FFFu + ((u >> 16) & 1u);
    return u >> 16;
}

// ---------------- prep: x fp32 -> bf16 ----------------
__global__ __launch_bounds__(256) void cvt_x_bf16(const float* __restrict__ x,
                                                  uint16_t* __restrict__ xb, int n) {
    int i = blockIdx.x * 256 + threadIdx.x;
    if (i < n) xb[i] = (uint16_t)f2bf_rne(x[i]);
}

// ---------------- prep: xA = x @ lora_A.T  (64 x 16) ----------------
__global__ __launch_bounds__(256) void lora_xa(const float* __restrict__ x,
                                               const float* __restrict__ A,
                                               float* __restrict__ xA) {
    __shared__ float srow[IN_F];        // 32 KB: one batch row of x
    __shared__ float part[16][16];      // [stripe][r]
    const int b = blockIdx.x;
    for (int i = threadIdx.x; i < IN_F; i += 256)
        srow[i] = x[(size_t)b * IN_F + i];
    __syncthreads();
    const int r = threadIdx.x & 15;
    const int s = threadIdx.x >> 4;
    const float* Ar = A + (size_t)r * IN_F;
    float acc = 0.f;
    for (int k = s; k < IN_F; k += 16) acc += srow[k] * Ar[k];
    part[s][r] = acc;
    __syncthreads();
    if (threadIdx.x < 16) {
        float t = 0.f;
        #pragma unroll
        for (int s2 = 0; s2 < 16; ++s2) t += part[s2][threadIdx.x];
        xA[b * RRANK + threadIdx.x] = t;
    }
}

// ---------------- main: fused dequant + WMMA GEMM + LoRA epilogue ----------------
__global__ __launch_bounds__(256) void qlora_main(
    const uint8_t* __restrict__ packed,   // OUT x IN/2 (nvfp4, low nibble = even idx)
    const float*   __restrict__ scales,   // OUT x IN/GROUP
    const __bf16*  __restrict__ xb,       // BATCH x IN (bf16)
    const float*   __restrict__ xA,       // BATCH x R
    const float*   __restrict__ loraB,    // OUT x R
    float*         __restrict__ out)      // BATCH x OUT
{
    __shared__ __bf16 wtile[32 * LDSW];       // 16.5 KB dequantized W tile (N x K)
    __shared__ float  s_xA[BATCH * RRANK];    // 4 KB
    __shared__ float  s_lB[32 * RRANK];       // 2 KB

    const int tid = threadIdx.x;
    const int n0  = blockIdx.x * 32;

    for (int i = tid; i < BATCH * RRANK; i += 256) s_xA[i] = xA[i];
    for (int i = tid; i < 32 * RRANK;   i += 256) s_lB[i] = loraB[(size_t)n0 * RRANK + i];

    const int wave = tid >> 5;                // 0..7
    const int lane = tid & 31;
    const int m0   = (wave & 3) * 16;         // M sub-tile (batch rows)
    const int nsub = (wave >> 2) * 16;        // N sub-tile within block
    const int half = lane >> 4;               // 0 | 1
    const int l16  = lane & 15;

    v8f acc = {};
    const __bf16* xrow = xb + (size_t)(m0 + l16) * IN_F;

    // v_perm selector constants (all byte indices 0-7 pick data, 0x0C = zero byte)
    const uint32_t SEL_A = 0x05010400u;   // interleave src1/src0 low-byte pairs
    const uint32_t SEL_B = 0x07030602u;   // interleave src1/src0 high-byte pairs
    // bf16 magnitude tables for nvfp4 {0,.5,1,1.5,2,3,4,6}: hi / lo bytes, mags 0-3 / 4-7
    const uint32_t TH_LO = 0x3F3F3F00u, TH_HI = 0x40404040u;
    const uint32_t TL_LO = 0xC0800000u, TL_HI = 0xC0804000u;

    for (int kc = 0; kc < IN_F; kc += KC) {
        __syncthreads();
        // -------- cooperative dequant: 32 rows x KC cols = 1024 u32 words --------
        #pragma unroll
        for (int j = 0; j < 4; ++j) {
            const int idx = tid + j * 256;
            const int row = idx >> 5;         // 0..31
            const int cw  = idx & 31;         // u32 word within chunk row
            const uint32_t* prow =
                (const uint32_t*)(packed + (size_t)(n0 + row) * (IN_F / 2));
            if (j == 0 && kc + KC < IN_F)     // prefetch next chunk's packed words
                __builtin_prefetch(&prow[((kc + KC) >> 3) + cw], 0, 3);
            const uint32_t w = prow[(kc >> 3) + cw];
            const float sc = scales[(size_t)(n0 + row) * (IN_F / GROUP)
                                    + (kc >> 4) + (cw >> 1)];
            const uint32_t scb = f2bf_rne(sc);
            const uint32_t sc2 = scb | (scb << 16);   // packed bf16 {sc, sc}

            // byte-parallel nvfp4 -> packed bf16 magnitude via v_perm table lookup
            const uint32_t e = w & 0x07070707u;          // mags of even elems (byte j)
            const uint32_t o = (w >> 4) & 0x07070707u;   // mags of odd elems
            const uint32_t sel01 = __builtin_amdgcn_perm(o, e, SEL_A); // mags 0..3
            const uint32_t sel23 = __builtin_amdgcn_perm(o, e, SEL_B); // mags 4..7
            const uint32_t hi01 = __builtin_amdgcn_perm(TH_HI, TH_LO, sel01);
            const uint32_t lo01 = __builtin_amdgcn_perm(TL_HI, TL_LO, sel01);
            const uint32_t hi23 = __builtin_amdgcn_perm(TH_HI, TH_LO, sel23);
            const uint32_t lo23 = __builtin_amdgcn_perm(TL_HI, TL_LO, sel23);
            uint32_t m0w = __builtin_amdgcn_perm(hi01, lo01, SEL_A);  // bf16 e0,e1
            uint32_t m1w = __builtin_amdgcn_perm(hi01, lo01, SEL_B);  // bf16 e2,e3
            uint32_t m2w = __builtin_amdgcn_perm(hi23, lo23, SEL_A);  // bf16 e4,e5
            uint32_t m3w = __builtin_amdgcn_perm(hi23, lo23, SEL_B);  // bf16 e6,e7

            // sign injection: even signs -> bit15, odd signs -> bit31 of each pair
            const uint32_t tO = w & 0x80808080u;          // odd-elem signs at byte.bit7
            const uint32_t tE = (w << 4) & 0x80808080u;   // even-elem signs at byte.bit7
            m0w ^= __builtin_amdgcn_perm(tO, tE, 0x040C000Cu);
            m1w ^= __builtin_amdgcn_perm(tO, tE, 0x050C010Cu);
            m2w ^= __builtin_amdgcn_perm(tO, tE, 0x060C020Cu);
            m3w ^= __builtin_amdgcn_perm(tO, tE, 0x070C030Cu);

            // scale with packed bf16 multiply (CDNA5 VOP3P)
            uint4 ov;
            asm("v_pk_mul_bf16 %0, %1, %2" : "=v"(ov.x) : "v"(m0w), "v"(sc2));
            asm("v_pk_mul_bf16 %0, %1, %2" : "=v"(ov.y) : "v"(m1w), "v"(sc2));
            asm("v_pk_mul_bf16 %0, %1, %2" : "=v"(ov.z) : "v"(m2w), "v"(sc2));
            asm("v_pk_mul_bf16 %0, %1, %2" : "=v"(ov.w) : "v"(m3w), "v"(sc2));
            *(uint4*)&wtile[row * LDSW + cw * 8] = ov;
        }
        __syncthreads();

        // -------- 8 WMMA steps over this K chunk --------
        #pragma unroll
        for (int ks = 0; ks < KC; ks += 32) {
            union { v16bf v; uint4 q[2]; } afrag, bfrag;
            const int k = kc + ks;
            // A 16x32 bf16 fragment: V0-3 <- K 8*half..+7, V4-7 <- K 16+8*half..+7
            afrag.q[0] = *(const uint4*)(xrow + k + 8 * half);
            afrag.q[1] = *(const uint4*)(xrow + k + 16 + 8 * half);
            // B 32x16 bf16 fragment: lane=col, V0-3 <- K 8*half..+7, V4-7 <- K 16+8*half..+7 (LDS)
            const __bf16* brow = &wtile[(nsub + l16) * LDSW + ks + 8 * half];
            bfrag.q[0] = *(const uint4*)(brow);
            bfrag.q[1] = *(const uint4*)(brow + 16);
            acc = __builtin_amdgcn_wmma_f32_16x16x32_bf16(
                false, afrag.v, false, bfrag.v, (short)0, acc, false, false);
        }
    }

    // -------- epilogue: LoRA rank-16 add + store --------
    #pragma unroll
    for (int v = 0; v < 8; ++v) {
        const int m = m0 + v + 8 * half;      // C frag: M = v + 8*(lane>=16)
        const int n = nsub + l16;             //         N = lane % 16
        float r = acc[v];
        #pragma unroll
        for (int j = 0; j < RRANK; ++j)
            r += s_xA[m * RRANK + j] * s_lB[n * RRANK + j];
        out[(size_t)m * OUT_F + n0 + n] = r;
    }
}

extern "C" void kernel_launch(void* const* d_in, const int* in_sizes, int n_in,
                              void* d_out, int out_size, void* d_ws, size_t ws_size,
                              hipStream_t stream) {
    (void)in_sizes; (void)n_in; (void)out_size; (void)ws_size;
    const float*   x      = (const float*)d_in[0];
    const uint8_t* packed = (const uint8_t*)d_in[1];
    const float*   scales = (const float*)d_in[2];
    const float*   loraA  = (const float*)d_in[3];
    const float*   loraB  = (const float*)d_in[4];
    float*         out    = (float*)d_out;

    // workspace: [ x_bf16 : 64*8192*2 B ][ xA : 64*16*4 B ]
    uint16_t* xbw = (uint16_t*)d_ws;
    float*    xAw = (float*)((char*)d_ws + (size_t)BATCH * IN_F * sizeof(uint16_t));

    const int nconv = BATCH * IN_F;
    cvt_x_bf16<<<(nconv + 255) / 256, 256, 0, stream>>>(x, xbw, nconv);
    lora_xa<<<BATCH, 256, 0, stream>>>(x, loraA, xAw);
    qlora_main<<<OUT_F / 32, 256, 0, stream>>>(packed, scales, (const __bf16*)xbw,
                                               xAw, loraB, out);
}
